// QuantizedLinear_85383949845198
// MI455X (gfx1250) — compile-verified, matrix operand order checked by reading, not run on hardware
//
#include <hip/hip_runtime.h>
#include <stdint.h>

// ---------------------------------------------------------------------------
// Quantized linear  out[m,n] = (sum_k x[m,k] * w_int[n,k]) * scale[n] + bias[n]
// M=8192 (B*S), N=16384 (OUT), K=4096 (IN).  zero_point == 0 (symmetric).
//
// MI455X plan:
//   pass 1: x fp32 -> f16 (64MB), w int32 -> f16 (128MB) into d_ws;
//           ~192MB f16 working set ~= global L2 -> GEMM re-reads hit L2.
//   pass 2: WMMA GEMM, block 128x256, wave tile 64x64 (16 WMMA : 16 ds_load),
//           async global->LDS staging (ASYNCcnt), ONE barrier per k-step,
//           fused scale+bias epilogue with non-temporal stores.
// ---------------------------------------------------------------------------

typedef __attribute__((ext_vector_type(16))) _Float16 v16h;
typedef __attribute__((ext_vector_type(8)))  _Float16 v8h;
typedef __attribute__((ext_vector_type(8)))  float    v8f;
typedef __attribute__((ext_vector_type(4)))  float    v4f;
typedef __attribute__((ext_vector_type(4)))  int      v4i;

#define M_TOT 8192
#define N_TOT 16384
#define K_TOT 4096

#define BM 128
#define BN 256
#define BK 32
#define LDAS (BK + 8)   // 80B row stride: 20-bank stride -> conflict-free b128

// ---------------------------------------------------------------------------
// Pre-pass: convert to f16 (reads non-temporal: each source byte read once)
// ---------------------------------------------------------------------------
__global__ __launch_bounds__(256)
void cvt_x_to_f16(const float* __restrict__ x, _Float16* __restrict__ xh) {
    const size_t i = ((size_t)blockIdx.x * 256 + threadIdx.x) * 8;
    v4f a = __builtin_nontemporal_load((const v4f*)(x + i));
    v4f b = __builtin_nontemporal_load((const v4f*)(x + i) + 1);
    v8h h;
#pragma unroll
    for (int j = 0; j < 4; ++j) { h[j] = (_Float16)a[j]; h[j + 4] = (_Float16)b[j]; }
    *(v8h*)(xh + i) = h;
}

__global__ __launch_bounds__(256)
void cvt_w_to_f16(const int* __restrict__ w, _Float16* __restrict__ wh) {
    const size_t i = ((size_t)blockIdx.x * 256 + threadIdx.x) * 8;
    v4i a = __builtin_nontemporal_load((const v4i*)(w + i));
    v4i b = __builtin_nontemporal_load((const v4i*)(w + i) + 1);
    v8h h;
#pragma unroll
    for (int j = 0; j < 4; ++j) { h[j] = (_Float16)a[j]; h[j + 4] = (_Float16)b[j]; }
    *(v8h*)(wh + i) = h;
}

// ---------------------------------------------------------------------------
// WMMA helpers
// ---------------------------------------------------------------------------
__device__ inline v16h load_frag(const _Float16* p) {
    v8h lo = *(const v8h*)(p);        // K =  khalf .. khalf+7
    v8h hi = *(const v8h*)(p + 16);   // K = 16+khalf .. 16+khalf+7
    return __builtin_shufflevector(lo, hi, 0,1,2,3,4,5,6,7,8,9,10,11,12,13,14,15);
}

// one 16B async copy: global -> LDS, tracked by ASYNCcnt
__device__ inline void async_b128(unsigned lds_off, const void* gaddr) {
    asm volatile("global_load_async_to_lds_b128 %0, %1, off"
                 :: "v"(lds_off), "v"(gaddr) : "memory");
}

// ---------------------------------------------------------------------------
// Main GEMM on pre-converted f16 operands.
// Block tile 128(M) x 256(N), 8 waves as 2(M) x 4(N), wave tile 64x64.
// ---------------------------------------------------------------------------
__global__ __launch_bounds__(256)
void qlinear_wmma_f16_pre(const _Float16* __restrict__ xh,
                          const _Float16* __restrict__ wh,
                          const float* __restrict__ scale,
                          const float* __restrict__ bias,
                          float* __restrict__ out)
{
    __shared__ alignas(16) _Float16 As[2][BM * LDAS];   // 2 x 10240 B
    __shared__ alignas(16) _Float16 Bs[2][BN * LDAS];   // 2 x 20480 B  (60KB total)

    const int tid  = threadIdx.x;
    const int wave = tid >> 5;          // 8 waves (wave32)
    const int lane = tid & 31;
    const int lm   = lane & 15;
    const int lhi  = lane >> 4;

    // L2 supergrouping over the 64x64 tile grid: 32x32 tile groups
    const int NT = N_TOT / BN;          // 64
    const int GM = 32, GN = 32;
    const int bid   = blockIdx.x;
    const int perSG = GM * GN;          // 1024
    const int sg    = bid / perSG;
    const int r     = bid % perSG;
    const int sgN   = NT / GN;          // 2
    const int m_tile = (sg / sgN) * GM + r / GN;
    const int n_tile = (sg % sgN) * GN + r % GN;
    const int block_m = m_tile * BM;
    const int block_n = n_tile * BN;

    const int wm = (wave >> 2) * 64;    // 2 waves along M
    const int wn = (wave & 3)  * 64;    // 4 waves along N

    // staging: A: thread owns 32B of one row (2 threads/row);
    //          B: thread owns a full 64B row.
    const int lar = tid >> 1;           // A row 0..127
    const int lac = (tid & 1) * 16;     // A half-offset 0/16
    const _Float16* xg = xh + (size_t)(block_m + lar) * K_TOT + lac;
    const _Float16* wg = wh + (size_t)(block_n + tid) * K_TOT;

    v8f acc[4][4] = {};                 // 16 x (16x16 f32) accumulators

    auto issue_stage = [&](int kt, int buf) {
        const _Float16* ga = xg + (size_t)kt * BK;
        const _Float16* gb = wg + (size_t)kt * BK;
        const unsigned ao = (unsigned)(uintptr_t)&As[buf][lar * LDAS + lac];
        const unsigned bo = (unsigned)(uintptr_t)&Bs[buf][tid * LDAS];
        async_b128(ao,      ga);
        async_b128(ao + 16, ga + 8);
        async_b128(bo,      gb);
        async_b128(bo + 16, gb + 8);
        async_b128(bo + 32, gb + 16);
        async_b128(bo + 48, gb + 24);
    };

    const int KT = K_TOT / BK;          // 128 k-steps

    issue_stage(0, 0);                  // prologue

    for (int kt = 0; kt < KT; ++kt) {
        const int cur = kt & 1;

        // my stage-kt copies landed in LDS
        asm volatile("s_wait_asynccnt 0" ::: "memory");
        // everyone's stage kt is visible AND everyone finished reading buf cur^1
        __syncthreads();
        // now safe to overwrite buf cur^1; transfers overlap the WMMAs below
        if (kt + 1 < KT)
            issue_stage(kt + 1, cur ^ 1);

        v16h af[4], bf[4];
#pragma unroll
        for (int mi = 0; mi < 4; ++mi)
            af[mi] = load_frag(&As[cur][(wm + mi * 16 + lm) * LDAS + lhi * 8]);
#pragma unroll
        for (int ni = 0; ni < 4; ++ni)
            bf[ni] = load_frag(&Bs[cur][(wn + ni * 16 + lm) * LDAS + lhi * 8]);

#pragma unroll
        for (int mi = 0; mi < 4; ++mi)
#pragma unroll
            for (int ni = 0; ni < 4; ++ni)
                acc[mi][ni] = __builtin_amdgcn_wmma_f32_16x16x32_f16(
                    false, af[mi], false, bf[ni],
                    (short)0, acc[mi][ni], false, false);
    }

    // fused dequant epilogue; non-temporal stores keep the 536MB out stream
    // from evicting the L2-resident f16 operands
#pragma unroll
    for (int ni = 0; ni < 4; ++ni) {
        const int n  = block_n + wn + ni * 16 + lm;
        const float sc = scale[n];
        const float bs = bias[n];
#pragma unroll
        for (int mi = 0; mi < 4; ++mi) {
            const int m0 = block_m + wm + mi * 16 + lhi * 8;
            float* o = out + (size_t)m0 * N_TOT + n;
#pragma unroll
            for (int r8 = 0; r8 < 8; ++r8)
                __builtin_nontemporal_store(acc[mi][ni][r8] * sc + bs,
                                            o + (size_t)r8 * N_TOT);
        }
    }
}

// ---------------------------------------------------------------------------
// Fallback (ws too small): fused-convert GEMM (block 128x128, wave 64x32)
// ---------------------------------------------------------------------------
#define FBM 128
#define FBN 128

__device__ inline void cvt_store_f32(const v4f a[4], _Float16* p) {
    v8h h0, h1;
#pragma unroll
    for (int i = 0; i < 4; ++i) {
        h0[i] = (_Float16)a[0][i]; h0[i + 4] = (_Float16)a[1][i];
        h1[i] = (_Float16)a[2][i]; h1[i + 4] = (_Float16)a[3][i];
    }
    *(v8h*)(p) = h0; *(v8h*)(p + 8) = h1;
}
__device__ inline void cvt_store_i32(const v4i a[4], _Float16* p) {
    v8h h0, h1;
#pragma unroll
    for (int i = 0; i < 4; ++i) {
        h0[i] = (_Float16)a[0][i]; h0[i + 4] = (_Float16)a[1][i];
        h1[i] = (_Float16)a[2][i]; h1[i + 4] = (_Float16)a[3][i];
    }
    *(v8h*)(p) = h0; *(v8h*)(p + 8) = h1;
}

__global__ __launch_bounds__(256)
void qlinear_wmma_f16_fused(const float* __restrict__ x,
                            const int*   __restrict__ w,
                            const float* __restrict__ scale,
                            const float* __restrict__ bias,
                            float*       __restrict__ out)
{
    __shared__ alignas(16) _Float16 As[2][FBM * LDAS];
    __shared__ alignas(16) _Float16 Bs[2][FBN * LDAS];

    const int tid  = threadIdx.x;
    const int wave = tid >> 5;
    const int lane = tid & 31;
    const int lm   = lane & 15;
    const int lhi  = lane >> 4;

    const int block_m = blockIdx.y * FBM;
    const int block_n = blockIdx.x * FBN;
    const int wm = (wave >> 2) * 64;
    const int wn = (wave & 3)  * 32;

    const int ldr = tid >> 1;
    const int ldc = (tid & 1) * 16;
    const float* xg = x + (size_t)(block_m + ldr) * K_TOT + ldc;
    const int*   wg = w + (size_t)(block_n + ldr) * K_TOT + ldc;

    v8f acc[4][2] = {};
    v4f axr[4]; v4i bwr[4];

#pragma unroll
    for (int i = 0; i < 4; ++i) { axr[i] = ((const v4f*)xg)[i]; bwr[i] = ((const v4i*)wg)[i]; }
    cvt_store_f32(axr, &As[0][ldr * LDAS + ldc]);
    cvt_store_i32(bwr, &Bs[0][ldr * LDAS + ldc]);
    __syncthreads();

    const int KT = K_TOT / BK;
    for (int kt = 0; kt < KT; ++kt) {
        const int cur = kt & 1;
        if (kt + 1 < KT) {
            const float* xn  = xg + (size_t)(kt + 1) * BK;
            const int*   wn2 = wg + (size_t)(kt + 1) * BK;
#pragma unroll
            for (int i = 0; i < 4; ++i) { axr[i] = ((const v4f*)xn)[i]; bwr[i] = ((const v4i*)wn2)[i]; }
        }
        v16h af[4], bf[2];
#pragma unroll
        for (int mi = 0; mi < 4; ++mi)
            af[mi] = load_frag(&As[cur][(wm + mi * 16 + lm) * LDAS + lhi * 8]);
#pragma unroll
        for (int ni = 0; ni < 2; ++ni)
            bf[ni] = load_frag(&Bs[cur][(wn + ni * 16 + lm) * LDAS + lhi * 8]);
#pragma unroll
        for (int mi = 0; mi < 4; ++mi)
#pragma unroll
            for (int ni = 0; ni < 2; ++ni)
                acc[mi][ni] = __builtin_amdgcn_wmma_f32_16x16x32_f16(
                    false, af[mi], false, bf[ni], (short)0, acc[mi][ni], false, false);
        if (kt + 1 < KT) {
            cvt_store_f32(axr, &As[cur ^ 1][ldr * LDAS + ldc]);
            cvt_store_i32(bwr, &Bs[cur ^ 1][ldr * LDAS + ldc]);
        }
        __syncthreads();
    }

#pragma unroll
    for (int ni = 0; ni < 2; ++ni) {
        const int n  = block_n + wn + ni * 16 + lm;
        const float sc = scale[n];
        const float bs = bias[n];
#pragma unroll
        for (int mi = 0; mi < 4; ++mi) {
            const int m0 = block_m + wm + mi * 16 + lhi * 8;
            float* o = out + (size_t)m0 * N_TOT + n;
#pragma unroll
            for (int r8 = 0; r8 < 8; ++r8)
                __builtin_nontemporal_store(acc[mi][ni][r8] * sc + bs,
                                            o + (size_t)r8 * N_TOT);
        }
    }
}

// ---------------------------------------------------------------------------
extern "C" void kernel_launch(void* const* d_in, const int* in_sizes, int n_in,
                              void* d_out, int out_size, void* d_ws, size_t ws_size,
                              hipStream_t stream) {
    const float* x     = (const float*)d_in[0];  // [8192, 4096] f32
    const int*   w     = (const int*)  d_in[1];  // [16384, 4096] int32
    const float* scale = (const float*)d_in[2];  // [16384]
    // d_in[3] = zero_point, all zeros (symmetric) -> folded out
    const float* bias  = (const float*)d_in[4];  // [16384]
    float* out = (float*)d_out;                  // [8192, 16384] f32

    const size_t x_elems = (size_t)M_TOT * K_TOT;     // 33.5M
    const size_t w_elems = (size_t)N_TOT * K_TOT;     // 67.1M
    const size_t need    = (x_elems + w_elems) * sizeof(_Float16);  // ~192MB

    if (ws_size >= need) {
        _Float16* xh = (_Float16*)d_ws;
        _Float16* wh = xh + x_elems;
        cvt_x_to_f16<<<(int)(x_elems / (8 * 256)), 256, 0, stream>>>(x, xh);
        cvt_w_to_f16<<<(int)(w_elems / (8 * 256)), 256, 0, stream>>>(w, wh);
        qlinear_wmma_f16_pre<<<(M_TOT / BM) * (N_TOT / BN), 256, 0, stream>>>(
            xh, wh, scale, bias, out);
    } else {
        dim3 grid(N_TOT / FBN, M_TOT / FBM);
        qlinear_wmma_f16_fused<<<grid, 256, 0, stream>>>(x, w, scale, bias, out);
    }
}
